// TttBaseModule_21337397527119
// MI455X (gfx1250) — matrix operand order repaired and checked by reading, not compiled, not run on hardware
//
#include <hip/hip_runtime.h>

typedef __bf16 bf16_t;
typedef __attribute__((ext_vector_type(8)))  bf16_t v8bf;
typedef __attribute__((ext_vector_type(16))) bf16_t v16bf;
typedef __attribute__((ext_vector_type(8)))  float  v8f;

// Problem constants (match reference)
constexpr int Bq  = 4;
constexpr int Lq  = 2048;
constexpr int Dq  = 2048;
constexpr int NHq = 32;
constexpr int HDq = 64;
constexpr int Kq  = 16;
constexpr int Mtok = Bq * Lq;      // 8192 token rows
constexpr int NCH  = Lq / Kq;      // 128 chunks
constexpr float EPSc = 1e-6f;

// ---------------------------------------------------------------------------
// Async Global->LDS copy (CDNA5 GLOBAL_LOAD_ASYNC_TO_LDS_B128, ASYNCcnt).
// Builtin signature (from compiler diagnostic): arg0 = v4i AS1*, arg1 = AS3*,
// then two immediate ints (offset, cpol).
// Guarded: falls back to VGPR staging when the builtins are unavailable.
// ---------------------------------------------------------------------------
#if defined(__AMDGCN__) &&                                                  \
    __has_builtin(__builtin_amdgcn_global_load_async_to_lds_b128) &&        \
    __has_builtin(__builtin_amdgcn_s_wait_asynccnt)
#define TTT_ASYNC_COPY 1
#else
#define TTT_ASYNC_COPY 0
#endif

#if TTT_ASYNC_COPY
typedef int b128vec_t __attribute__((vector_size(16)));
typedef __attribute__((address_space(1))) b128vec_t* gptr128_t;
typedef __attribute__((address_space(3))) b128vec_t* lptr128_t;

static __device__ __forceinline__ void async_copy16(const bf16_t* g, bf16_t* l) {
  gptr128_t gp = (gptr128_t)(reinterpret_cast<uintptr_t>(g));
  lptr128_t lp = (lptr128_t)(static_cast<uint32_t>(reinterpret_cast<uintptr_t>(l)));
  __builtin_amdgcn_global_load_async_to_lds_b128(gp, lp, 0, 0);
}
#endif

// ---------------------------------------------------------------------------
// Helpers
// ---------------------------------------------------------------------------
static __device__ __forceinline__ v16bf combine8(const bf16_t* plo, const bf16_t* phi) {
  v8bf lo = *(const v8bf*)plo;
  v8bf hi = *(const v8bf*)phi;
  return __builtin_shufflevector(lo, hi, 0,1,2,3,4,5,6,7,8,9,10,11,12,13,14,15);
}

static __device__ __forceinline__ v8f wmma_bf16(v16bf a, v16bf b, v8f c) {
  // D = A(16x32) * B(32x16) + C, f32 accumulate
  return __builtin_amdgcn_wmma_f32_16x16x32_bf16(false, a, false, b, (short)0, c,
                                                 false, false);
}

// ---------------------------------------------------------------------------
// f32 -> bf16 elementwise convert
// ---------------------------------------------------------------------------
__global__ __launch_bounds__(256) void f32_to_bf16_kernel(const float* __restrict__ in,
                                                          bf16_t* __restrict__ out,
                                                          size_t n) {
  size_t i = (size_t)blockIdx.x * blockDim.x + threadIdx.x;
  size_t stride = (size_t)gridDim.x * blockDim.x;
  for (; i < n; i += stride) out[i] = (bf16_t)in[i];
}

// ---------------------------------------------------------------------------
// coeff[b,h,l] = (1/(k+1)) * sigmoid(x_token . lw[h] + lb[h]) / HD   (LR = 1)
// one block per token
// ---------------------------------------------------------------------------
__global__ __launch_bounds__(256) void coeff_kernel(const float* __restrict__ X,
                                                    const float* __restrict__ lw,
                                                    const float* __restrict__ lb,
                                                    float* __restrict__ coeff) {
  __shared__ float xrow[Dq];
  const int t = blockIdx.x;  // 0..Mtok-1
  for (int d = threadIdx.x; d < Dq; d += 256) xrow[d] = X[(size_t)t * Dq + d];
  __syncthreads();
  const int wave = threadIdx.x >> 5, lane = threadIdx.x & 31;
  const int bb = t / Lq, li = t % Lq, k = li % Kq;
  for (int h = wave; h < NHq; h += 8) {
    float s = 0.f;
    for (int d = lane; d < Dq; d += 32) s += xrow[d] * lw[(size_t)h * Dq + d];
#pragma unroll
    for (int m = 16; m >= 1; m >>= 1) s += __shfl_xor(s, m, 32);
    if (lane == 0) {
      float v   = s + lb[h];
      float sig = 1.f / (1.f + __expf(-v));
      coeff[((size_t)(bb * NHq + h)) * Lq + li] =
          (1.0f / (float)(k + 1)) * sig * (1.f / (float)HDq);
    }
  }
}

// ---------------------------------------------------------------------------
// C[M,N] = A[M,K] * Bw[N,K]^T  (i.e. X @ W.T) — bf16 in, f32 out, WMMA.
// Block: 256 thr = 8 waves, tile 128(M) x 128(N), K stepped by 32.
// Double-buffered LDS; tiles staged with async Global->LDS copies when
// available (s_wait_asynccnt + barrier), else VGPR staging.
// Wave (wm = w>>2 in 0..1, wn = w&3 in 0..3) owns a 64x32 patch = 4x2 tiles.
// ---------------------------------------------------------------------------
__global__ __launch_bounds__(256) void gemm_bf16_nt(const bf16_t* __restrict__ A,
                                                    const bf16_t* __restrict__ Bw,
                                                    float* __restrict__ C,
                                                    int M, int N, int Kd) {
  __shared__ __align__(16) bf16_t As[2][128][32];  // [buf][m][k]
  __shared__ __align__(16) bf16_t Bs[2][128][32];  // [buf][n][k] (== W rows)
  const int tid = threadIdx.x, lane = tid & 31, wave = tid >> 5;
  const int wm = wave >> 2, wn = wave & 3;
  const int m0 = blockIdx.y * 128, n0 = blockIdx.x * 128;
  const int r = lane & 15, hh = lane >> 4;

  // Per-thread staging coordinates: 512 16B chunks per tile, 2 per thread.
  const int c0 = tid, c1 = tid + 256;
  const int row0 = c0 >> 2, col0 = c0 & 3;
  const int row1 = c1 >> 2, col1 = c1 & 3;
  const bf16_t* gA0 = &A[(size_t)(m0 + row0) * Kd + col0 * 8];
  const bf16_t* gA1 = &A[(size_t)(m0 + row1) * Kd + col1 * 8];
  const bf16_t* gB0 = &Bw[(size_t)(n0 + row0) * Kd + col0 * 8];
  const bf16_t* gB1 = &Bw[(size_t)(n0 + row1) * Kd + col1 * 8];

  v8f acc[4][2];
#pragma unroll
  for (int mt = 0; mt < 4; ++mt)
#pragma unroll
    for (int nt = 0; nt < 2; ++nt) acc[mt][nt] = (v8f){0.f,0.f,0.f,0.f,0.f,0.f,0.f,0.f};

#if TTT_ASYNC_COPY
  // prologue: async-stage k-tile 0 into buffer 0
  async_copy16(gA0, &As[0][row0][col0 * 8]);
  async_copy16(gA1, &As[0][row1][col1 * 8]);
  async_copy16(gB0, &Bs[0][row0][col0 * 8]);
  async_copy16(gB1, &Bs[0][row1][col1 * 8]);
#endif

  int p = 0;
  for (int k0 = 0; k0 < Kd; k0 += 32, p ^= 1) {
#if TTT_ASYNC_COPY
    __builtin_amdgcn_s_wait_asynccnt(0);  // my async writes for buffer p done
    __syncthreads();                      // everyone's writes visible
    if (k0 + 32 < Kd) {
      const int q = p ^ 1;
      async_copy16(gA0 + (k0 + 32), &As[q][row0][col0 * 8]);
      async_copy16(gA1 + (k0 + 32), &As[q][row1][col1 * 8]);
      async_copy16(gB0 + (k0 + 32), &Bs[q][row0][col0 * 8]);
      async_copy16(gB1 + (k0 + 32), &Bs[q][row1][col1 * 8]);
    }
#else
    *(uint4*)&As[p][row0][col0 * 8] = *(const uint4*)(gA0 + k0);
    *(uint4*)&As[p][row1][col1 * 8] = *(const uint4*)(gA1 + k0);
    *(uint4*)&Bs[p][row0][col0 * 8] = *(const uint4*)(gB0 + k0);
    *(uint4*)&Bs[p][row1][col1 * 8] = *(const uint4*)(gB1 + k0);
    __syncthreads();
#endif

    v16bf af[4], bfr[2];
#pragma unroll
    for (int mt = 0; mt < 4; ++mt) {
      int ml = wm * 64 + mt * 16 + r;
      af[mt] = combine8(&As[p][ml][hh * 8], &As[p][ml][16 + hh * 8]);
    }
#pragma unroll
    for (int nt = 0; nt < 2; ++nt) {
      int nl = wn * 32 + nt * 16 + r;
      bfr[nt] = combine8(&Bs[p][nl][hh * 8], &Bs[p][nl][16 + hh * 8]);
    }
#pragma unroll
    for (int mt = 0; mt < 4; ++mt)
#pragma unroll
      for (int nt = 0; nt < 2; ++nt) acc[mt][nt] = wmma_bf16(af[mt], bfr[nt], acc[mt][nt]);

#if !TTT_ASYNC_COPY
    __syncthreads();
#endif
  }

#pragma unroll
  for (int mt = 0; mt < 4; ++mt)
#pragma unroll
    for (int nt = 0; nt < 2; ++nt)
#pragma unroll
      for (int j = 0; j < 8; ++j) {
        int row = m0 + wm * 64 + mt * 16 + hh * 8 + j;
        int col = n0 + wn * 32 + nt * 16 + r;
        C[(size_t)row * N + col] = acc[mt][nt][j];
      }
}

// ---------------------------------------------------------------------------
// TTT scan: one wave per (b,h). W1 (64x64 f32) lives in LDS across 128 chunks.
// All small matmuls via WMMA bf16 (K=16 contractions zero-padded to 32).
// grad is stored NEGATED in bf16 (gradT) so every use is a pure accumulate.
// Output written in place over XA.
// ---------------------------------------------------------------------------
__global__ __launch_bounds__(32) void ttt_scan(const float* __restrict__ XC,
                                               const float* __restrict__ XB,
                                               float* __restrict__ XA,  // in: V, out: XCW
                                               const float* __restrict__ coeff,
                                               const float* __restrict__ ln_w,
                                               const float* __restrict__ ln_b,
                                               const float* __restrict__ W1_init,
                                               const float* __restrict__ b1_init) {
  __shared__ float W1s[HDq][HDq];                    // [k_in][n_out], f32 master
  __shared__ __align__(16) bf16_t W1bT[HDq][HDq];    // [n_out][k_in], WMMA B layout
  __shared__ float b1s[HDq];
  __shared__ float xcs[Kq][HDq], xbs[Kq][HDq], xas[Kq][HDq];
  __shared__ __align__(16) bf16_t xcb[Kq][HDq], xbb[Kq][HDq];
  __shared__ __align__(16) bf16_t xbTs[HDq][32];     // A for W1 update: co[k]*xb[k][f]
  __shared__ float grads[Kq][HDq];
  __shared__ __align__(16) bf16_t gradT[HDq][32];    // B layout, = -grad, k-padded
  __shared__ __align__(16) bf16_t AEb[Kq][32];       // (Attn∘E + E), k-padded
  __shared__ float Attns[Kq][Kq];
  __shared__ float Z1s[Kq][HDq], Z1bs[Kq][HDq];
  __shared__ float cos_[Kq];

  const int lane = threadIdx.x;
  const int bh = blockIdx.x;
  const int b = bh / NHq, h = bh % NHq;
  const int r = lane & 15, hh = lane >> 4;

  const float gam0 = ln_w[h * HDq + lane], gam1 = ln_w[h * HDq + lane + 32];
  const float bet0 = ln_b[h * HDq + lane], bet1 = ln_b[h * HDq + lane + 32];

  // init state
  for (int id = lane; id < HDq * HDq; id += 32) {
    int k = id >> 6, nn = id & 63;
    float w = W1_init[(size_t)h * HDq * HDq + id];
    W1s[k][nn] = w;
    W1bT[nn][k] = (bf16_t)w;
  }
  b1s[lane]      = b1_init[h * HDq + lane];
  b1s[lane + 32] = b1_init[h * HDq + lane + 32];
  // zero the K-padding regions once (never overwritten afterwards)
  for (int id = lane; id < HDq * 16; id += 32) {
    int f = id >> 4, c = 16 + (id & 15);
    gradT[f][c] = (bf16_t)0.f;
    xbTs[f][c]  = (bf16_t)0.f;
  }
  for (int id = lane; id < Kq * 16; id += 32) {
    int i = id >> 4, c = 16 + (id & 15);
    AEb[i][c] = (bf16_t)0.f;
  }
  __syncthreads();

  for (int ch = 0; ch < NCH; ++ch) {
    const size_t rowbase = (size_t)b * Lq + (size_t)ch * Kq;
    // ---- load chunk tiles ----
    for (int id = lane; id < Kq * HDq; id += 32) {
      int t = id >> 6, f = id & 63;
      size_t gi = (rowbase + t) * Dq + (size_t)h * HDq + f;
      float vc = XC[gi], vb = XB[gi], va = XA[gi];
      xcs[t][f] = vc; xbs[t][f] = vb; xas[t][f] = va;
      xcb[t][f] = (bf16_t)vc; xbb[t][f] = (bf16_t)vb;
    }
    if (lane < Kq) cos_[lane] = coeff[(size_t)bh * Lq + ch * Kq + lane];
    __syncthreads();
    for (int id = lane; id < HDq * Kq; id += 32) {
      int f = id >> 4, k = id & 15;
      xbTs[f][k] = (bf16_t)(cos_[k] * xbs[k][f]);
    }

    // ---- Z1 = xb @ W1 + b1 ----
#pragma unroll
    for (int nt = 0; nt < 4; ++nt) {
      int col = nt * 16 + r;
      float bv = b1s[col];
      v8f acc = (v8f){bv, bv, bv, bv, bv, bv, bv, bv};
#pragma unroll
      for (int kb = 0; kb < 2; ++kb) {
        v16bf a = combine8(&xbb[r][kb * 32 + hh * 8], &xbb[r][kb * 32 + 16 + hh * 8]);
        v16bf bb = combine8(&W1bT[col][kb * 32 + hh * 8], &W1bT[col][kb * 32 + 16 + hh * 8]);
        acc = wmma_bf16(a, bb, acc);
      }
#pragma unroll
      for (int j = 0; j < 8; ++j) Z1s[hh * 8 + j][col] = acc[j];
    }
    __syncthreads();

    // ---- grad = ln_fused_l2_bwd(Z1, xa-xb); store -grad bf16 for WMMA ----
    for (int t = 0; t < Kq; ++t) {
      float z0 = Z1s[t][lane], z1 = Z1s[t][lane + 32];
      float s = z0 + z1, s2 = z0 * z0 + z1 * z1;
#pragma unroll
      for (int m = 16; m >= 1; m >>= 1) { s += __shfl_xor(s, m, 32); s2 += __shfl_xor(s2, m, 32); }
      float mu = s * (1.f / HDq);
      float var = s2 * (1.f / HDq) - mu * mu;
      float rstd = rsqrtf(var + EPSc);
      float xh0 = (z0 - mu) * rstd, xh1 = (z1 - mu) * rstd;
      float y0 = gam0 * xh0 + bet0, y1 = gam1 * xh1 + bet1;
      float tg0 = xas[t][lane] - xbs[t][lane];
      float tg1 = xas[t][lane + 32] - xbs[t][lane + 32];
      float gg0 = (y0 - tg0) * gam0, gg1 = (y1 - tg1) * gam1;
      float sg = gg0 + gg1, sgx = gg0 * xh0 + gg1 * xh1;
#pragma unroll
      for (int m = 16; m >= 1; m >>= 1) { sg += __shfl_xor(sg, m, 32); sgx += __shfl_xor(sgx, m, 32); }
      float gr0 = ((float)HDq * gg0 - sg - xh0 * sgx) * (rstd * (1.f / HDq));
      float gr1 = ((float)HDq * gg1 - sg - xh1 * sgx) * (rstd * (1.f / HDq));
      grads[t][lane] = gr0; grads[t][lane + 32] = gr1;
      gradT[lane][t] = (bf16_t)(-gr0); gradT[lane + 32][t] = (bf16_t)(-gr1);
    }
    __syncthreads();

    // ---- Attn = xc @ xb^T (raw; masking folded into AE) ----
    {
      v8f acc = (v8f){0.f,0.f,0.f,0.f,0.f,0.f,0.f,0.f};
#pragma unroll
      for (int kb = 0; kb < 2; ++kb) {
        v16bf a = combine8(&xcb[r][kb * 32 + hh * 8], &xcb[r][kb * 32 + 16 + hh * 8]);
        v16bf bb = combine8(&xbb[r][kb * 32 + hh * 8], &xbb[r][kb * 32 + 16 + hh * 8]);
        acc = wmma_bf16(a, bb, acc);
      }
#pragma unroll
      for (int j = 0; j < 8; ++j) Attns[hh * 8 + j][r] = acc[j];
    }
    __syncthreads();

    // ---- AE[i][j] = co[j]*(Attn[i][j]+1) for j<=i else 0 ----
    for (int id = lane; id < Kq * Kq; id += 32) {
      int i = id >> 4, j = id & 15;
      float v = (j <= i) ? cos_[j] * (Attns[i][j] + 1.f) : 0.f;
      AEb[i][j] = (bf16_t)v;
    }
    __syncthreads();

    // ---- Z1_bar = xc@W1 + b1 + AE@(-grad) ----
#pragma unroll
    for (int nt = 0; nt < 4; ++nt) {
      int col = nt * 16 + r;
      float bv = b1s[col];
      v8f acc = (v8f){bv, bv, bv, bv, bv, bv, bv, bv};
#pragma unroll
      for (int kb = 0; kb < 2; ++kb) {
        v16bf a = combine8(&xcb[r][kb * 32 + hh * 8], &xcb[r][kb * 32 + 16 + hh * 8]);
        v16bf bb = combine8(&W1bT[col][kb * 32 + hh * 8], &W1bT[col][kb * 32 + 16 + hh * 8]);
        acc = wmma_bf16(a, bb, acc);
      }
      {
        v16bf a = combine8(&AEb[r][hh * 8], &AEb[r][16 + hh * 8]);
        v16bf bb = combine8(&gradT[col][hh * 8], &gradT[col][16 + hh * 8]);
        acc = wmma_bf16(a, bb, acc);
      }
#pragma unroll
      for (int j = 0; j < 8; ++j) Z1bs[hh * 8 + j][col] = acc[j];
    }
    __syncthreads();

    // ---- out = xc + ln_fwd(Z1_bar); write over XA ----
    for (int t = 0; t < Kq; ++t) {
      float z0 = Z1bs[t][lane], z1 = Z1bs[t][lane + 32];
      float s = z0 + z1, s2 = z0 * z0 + z1 * z1;
#pragma unroll
      for (int m = 16; m >= 1; m >>= 1) { s += __shfl_xor(s, m, 32); s2 += __shfl_xor(s2, m, 32); }
      float mu = s * (1.f / HDq);
      float var = s2 * (1.f / HDq) - mu * mu;
      float rstd = rsqrtf(var + EPSc);
      float o0 = xcs[t][lane] + gam0 * ((z0 - mu) * rstd) + bet0;
      float o1 = xcs[t][lane + 32] + gam1 * ((z1 - mu) * rstd) + bet1;
      size_t gi = (rowbase + t) * Dq + (size_t)h * HDq;
      XA[gi + lane] = o0;
      XA[gi + lane + 32] = o1;
    }
    // ---- b1 -= sum_k co[k]*grad[k][:] ----
    {
      float s0 = 0.f, s1 = 0.f;
      for (int k = 0; k < Kq; ++k) {
        s0 += cos_[k] * grads[k][lane];
        s1 += cos_[k] * grads[k][lane + 32];
      }
      b1s[lane] -= s0;
      b1s[lane + 32] -= s1;
    }
    __syncthreads();

    // ---- W1 += (co*xb)^T @ (-grad) ----
#pragma unroll
    for (int mt = 0; mt < 4; ++mt) {
      v16bf a = combine8(&xbTs[mt * 16 + r][hh * 8], &xbTs[mt * 16 + r][16 + hh * 8]);
#pragma unroll
      for (int nt = 0; nt < 4; ++nt) {
        int col = nt * 16 + r;
        v8f acc;
#pragma unroll
        for (int j = 0; j < 8; ++j) acc[j] = W1s[mt * 16 + hh * 8 + j][col];
        v16bf bb = combine8(&gradT[col][hh * 8], &gradT[col][16 + hh * 8]);
        acc = wmma_bf16(a, bb, acc);
#pragma unroll
        for (int j = 0; j < 8; ++j) {
          int rowf = mt * 16 + hh * 8 + j;
          W1s[rowf][col] = acc[j];
          W1bT[col][rowf] = (bf16_t)acc[j];
        }
      }
    }
    __syncthreads();
  }
}

// ---------------------------------------------------------------------------
// launch
// ---------------------------------------------------------------------------
extern "C" void kernel_launch(void* const* d_in, const int* in_sizes, int n_in,
                              void* d_out, int out_size, void* d_ws, size_t ws_size,
                              hipStream_t stream) {
  (void)in_sizes; (void)n_in; (void)out_size; (void)ws_size;
  const float* hidden = (const float*)d_in[0];
  const float* Wq = (const float*)d_in[1];
  const float* Wk = (const float*)d_in[2];
  const float* Wv = (const float*)d_in[3];
  const float* Wo = (const float*)d_in[4];
  const float* lw = (const float*)d_in[5];
  const float* lb = (const float*)d_in[6];
  const float* lnw = (const float*)d_in[7];
  const float* lnb = (const float*)d_in[8];
  const float* W1i = (const float*)d_in[9];
  const float* b1i = (const float*)d_in[10];
  float* out = (float*)d_out;

  char* ws = (char*)d_ws;
  size_t off = 0;
  auto carve = [&](size_t bytes) -> void* {
    void* p = ws + off;
    off += (bytes + 255) & ~(size_t)255;
    return p;
  };
  bf16_t* Xbf = (bf16_t*)carve((size_t)Mtok * Dq * 2);  // reused for XCW bf16
  bf16_t* Wqb = (bf16_t*)carve((size_t)Dq * Dq * 2);
  bf16_t* Wkb = (bf16_t*)carve((size_t)Dq * Dq * 2);
  bf16_t* Wvb = (bf16_t*)carve((size_t)Dq * Dq * 2);
  bf16_t* Wob = (bf16_t*)carve((size_t)Dq * Dq * 2);
  float* XCb = (float*)carve((size_t)Mtok * Dq * 4);
  float* XBb = (float*)carve((size_t)Mtok * Dq * 4);
  float* XAb = (float*)carve((size_t)Mtok * Dq * 4);   // scan output in-place
  float* coeff = (float*)carve((size_t)Bq * NHq * Lq * 4);

  f32_to_bf16_kernel<<<4096, 256, 0, stream>>>(hidden, Xbf, (size_t)Mtok * Dq);
  f32_to_bf16_kernel<<<2048, 256, 0, stream>>>(Wq, Wqb, (size_t)Dq * Dq);
  f32_to_bf16_kernel<<<2048, 256, 0, stream>>>(Wk, Wkb, (size_t)Dq * Dq);
  f32_to_bf16_kernel<<<2048, 256, 0, stream>>>(Wv, Wvb, (size_t)Dq * Dq);
  f32_to_bf16_kernel<<<2048, 256, 0, stream>>>(Wo, Wob, (size_t)Dq * Dq);

  coeff_kernel<<<Mtok, 256, 0, stream>>>(hidden, lw, lb, coeff);

  dim3 gg(Dq / 128, Mtok / 128);  // (16, 64)
  gemm_bf16_nt<<<gg, 256, 0, stream>>>(Xbf, Wqb, XCb, Mtok, Dq, Dq);
  gemm_bf16_nt<<<gg, 256, 0, stream>>>(Xbf, Wkb, XBb, Mtok, Dq, Dq);
  gemm_bf16_nt<<<gg, 256, 0, stream>>>(Xbf, Wvb, XAb, Mtok, Dq, Dq);

  ttt_scan<<<Bq * NHq, 32, 0, stream>>>(XCb, XBb, XAb, coeff, lnw, lnb, W1i, b1i);

  f32_to_bf16_kernel<<<4096, 256, 0, stream>>>(XAb, Xbf, (size_t)Mtok * Dq);
  gemm_bf16_nt<<<gg, 256, 0, stream>>>(Xbf, Wob, out, Mtok, Dq, Dq);
}
